// GCN_61065845014917
// MI455X (gfx1250) — matrix operand order repaired and checked by reading, not compiled, not run on hardware
//
#include <hip/hip_runtime.h>
#include <math.h>

#define B_    4
#define C_    1024
#define D_    768
#define H_    12
#define E_    32
#define K_    2
#define S_    24
#define V_    8
#define P_    512
#define NT_   256
#define EMB_  768
#define BLK_  64
#define NL_   97
#define NLPAD_ 112
#define KIN_  1792          // 2*D + NT (concat input dim)
#define KBIL_ 49152         // EMB * BLK
#define MT_   4             // M-tiles per wave (register blocking)

typedef _Float16 h16;
typedef __attribute__((ext_vector_type(16))) _Float16 v16h;
typedef __attribute__((ext_vector_type(8)))  _Float16 v8h;
typedef __attribute__((ext_vector_type(8)))  float    v8f;

// ---------------- WMMA helpers (CDNA5 wave32, 16x16x32 f16 -> f32) ----------

__device__ __forceinline__ v8f wmma16(v16h a, v16h b, v8f c) {
  // (neg_a, A, neg_b, B, c_mod, C, reuse_a, reuse_b)
  return __builtin_amdgcn_wmma_f32_16x16x32_f16(false, a, false, b, (short)0, c,
                                                false, false);
}

// A fragment: 16x32 f16, row = lane%16.
// lane<16 : e[0..15]=A[row,kb+0..15]
// lane>=16: e[0..15]=A[row,kb+16..31]
__device__ __forceinline__ v16h a_frag(const h16* rowp, int kb, int lane) {
  int off = kb + ((lane & 16) ? 16 : 0);
  v8h lo = *(const v8h*)(rowp + off);
  v8h hi = *(const v8h*)(rowp + off + 8);
  return __builtin_shufflevector(lo, hi, 0, 1, 2, 3, 4, 5, 6, 7,
                                 8, 9, 10, 11, 12, 13, 14, 15);
}

// B fragment: 32x16 f16, col = lane%16; lane<16 holds K=kb..kb+15,
// lane>=16 holds K=kb+16..kb+31. colp points at a K-contiguous (transposed)
// weight column, so this is one aligned 32-byte load.
__device__ __forceinline__ v16h b_frag(const h16* colp, int kb, int lane) {
  return *(const v16h*)(colp + kb + ((lane & 16) ? 16 : 0));
}

// ---------------- prep kernels ----------------------------------------------

// Wt[n*K + k] = f16(W[k*N + n]); rows n in [N, Npad) zeroed.
__global__ void transpose_w_kernel(const float* __restrict__ W, h16* __restrict__ Wt,
                                   int K, int N, int Npad) {
  size_t total = (size_t)Npad * (size_t)K;
  for (size_t idx = (size_t)blockIdx.x * blockDim.x + threadIdx.x; idx < total;
       idx += (size_t)gridDim.x * blockDim.x) {
    int n = (int)(idx / K);
    int k = (int)(idx % K);
    Wt[idx] = (n < N) ? (h16)W[(size_t)k * N + n] : (h16)0.0f;
  }
}

// seqT[(b*D + d)*C + c] = f16(seq[(b*C + c)*D + d])
__global__ void seq_transpose_kernel(const float* __restrict__ seq,
                                     h16* __restrict__ seqT) {
  size_t total = (size_t)B_ * C_ * D_;
  for (size_t idx = (size_t)blockIdx.x * blockDim.x + threadIdx.x; idx < total;
       idx += (size_t)gridDim.x * blockDim.x) {
    int b = (int)(idx / ((size_t)C_ * D_));
    int rem = (int)(idx % ((size_t)C_ * D_));
    int d = rem / C_;
    int c = rem % C_;
    seqT[idx] = (h16)seq[((size_t)b * C_ + c) * D_ + d];
  }
}

// mention / sent / virtual row gathers straight into d_out (f32).
__global__ __launch_bounds__(256) void gather_kernel(
    const float* __restrict__ seq, const int* __restrict__ entity_pos,
    const int* __restrict__ sent_pos, const int* __restrict__ virtual_pos,
    float* __restrict__ out_mention, float* __restrict__ out_sent,
    float* __restrict__ out_virt) {
  const int MROWS = B_ * E_ * K_;   // 256
  const int SROWS = B_ * S_;        // 96
  int r = blockIdx.x;
  int b, tok;
  float* dst;
  if (r < MROWS) {
    b = r / (E_ * K_);
    tok = entity_pos[r] + 1;
    dst = out_mention + (size_t)r * D_;
  } else if (r < MROWS + SROWS) {
    int rr = r - MROWS;
    b = rr / S_;
    tok = sent_pos[rr] + 1;
    dst = out_sent + (size_t)rr * D_;
  } else {
    int rr = r - MROWS - SROWS;
    b = rr / V_;
    tok = virtual_pos[rr] + 1;
    dst = out_virt + (size_t)rr * D_;
  }
  const float* src = seq + ((size_t)b * C_ + tok) * D_;
  for (int d = threadIdx.x; d < D_; d += 256) dst[d] = src[d];
}

// logsumexp over K=2 mentions + node-type concat -> ent_hidden f16 [B*E][1024]
__global__ __launch_bounds__(256) void entity_kernel(
    const float* __restrict__ mention, const float* __restrict__ node_type,
    h16* __restrict__ ent_hidden) {
  int be = blockIdx.x;                               // 0..B*E-1
  const float* m0 = mention + (size_t)be * K_ * D_;  // row (b, e*K+0)
  const float* m1 = m0 + D_;
  h16* out = ent_hidden + (size_t)be * (D_ + NT_);
  for (int d = threadIdx.x; d < D_; d += 256) {
    float a = m0[d], c = m1[d];
    float mx = fmaxf(a, c);
    out[d] = (h16)(mx + logf(expf(a - mx) + expf(c - mx)));
  }
  for (int t = threadIdx.x; t < NT_; t += 256)
    out[D_ + t] = (h16)node_type[(size_t)be * NT_ + t];
}

// e_att[b,e,h,c] = mean_k attention[b,h,pos(b,e,k)+1,c]
__global__ __launch_bounds__(256) void eatt_kernel(
    const float* __restrict__ att, const int* __restrict__ entity_pos,
    float* __restrict__ e_att) {
  int beh = blockIdx.x;              // 0..B*E*H-1
  int h = beh % H_;
  int be = beh / H_;
  int b = be / E_;
  const float* base = att + (size_t)(b * H_ + h) * C_ * C_;
  const float* r0 = base + (size_t)(entity_pos[be * K_ + 0] + 1) * C_;
  const float* r1 = base + (size_t)(entity_pos[be * K_ + 1] + 1) * C_;
  float* o = e_att + ((size_t)be * H_ + h) * C_;
  for (int c = threadIdx.x; c < C_; c += 256) o[c] = 0.5f * (r0[c] + r1[c]);
}

// ht_att: product over heads, mean, normalize; emit f16 [B*P][C]
__global__ __launch_bounds__(256) void htatt_kernel(
    const float* __restrict__ e_att, const int* __restrict__ hts,
    h16* __restrict__ htatt_h) {
  int bp = blockIdx.x;               // 0..B*P-1
  int b = bp / P_;
  int hidx = hts[(size_t)bp * 2 + 0];
  int tidx = hts[(size_t)bp * 2 + 1];
  const float* eh = e_att + (size_t)(b * E_ + hidx) * H_ * C_;
  const float* et = e_att + (size_t)(b * E_ + tidx) * H_ * C_;
  float acc[4] = {0.f, 0.f, 0.f, 0.f};
  for (int hd = 0; hd < H_; ++hd) {
    const float* p0 = eh + (size_t)hd * C_;
    const float* p1 = et + (size_t)hd * C_;
#pragma unroll
    for (int q = 0; q < 4; ++q) {
      int c = threadIdx.x + q * 256;
      acc[q] += p0[c] * p1[c];
    }
  }
  float psum = 0.f;
#pragma unroll
  for (int q = 0; q < 4; ++q) { acc[q] *= (1.0f / H_); psum += acc[q]; }
  __shared__ float red[256];
  red[threadIdx.x] = psum;
  __syncthreads();
  for (int s = 128; s > 0; s >>= 1) {
    if (threadIdx.x < s) red[threadIdx.x] += red[threadIdx.x + s];
    __syncthreads();
  }
  float inv = 1.0f / (red[0] + 1e-5f);
  h16* o = htatt_h + (size_t)bp * C_;
#pragma unroll
  for (int q = 0; q < 4; ++q) {
    int c = threadIdx.x + q * 256;
    o[c] = (h16)(acc[q] * inv);
  }
}

// headin/tailin[:, 0:1024) = ent_hidden rows selected by hts
__global__ __launch_bounds__(256) void pairgather_kernel(
    const h16* __restrict__ ent_hidden, const int* __restrict__ hts,
    h16* __restrict__ headin, h16* __restrict__ tailin) {
  int bp = blockIdx.x;
  int b = bp / P_;
  const h16* hr = ent_hidden + (size_t)(b * E_ + hts[(size_t)bp * 2 + 0]) * (D_ + NT_);
  const h16* tr = ent_hidden + (size_t)(b * E_ + hts[(size_t)bp * 2 + 1]) * (D_ + NT_);
  h16* ho = headin + (size_t)bp * KIN_;
  h16* to = tailin + (size_t)bp * KIN_;
  for (int d = threadIdx.x; d < (D_ + NT_); d += 256) {
    ho[d] = hr[d];
    to[d] = tr[d];
  }
}

// ---------------- WMMA GEMM 1: rs = ht_att @ seq ----------------------------
// Per batch: [P x C] x [C x D]. One wave computes MT_ (=4) 16x16 M-tiles for
// one N-tile: each B fragment is loaded once and fed to 4 independent WMMAs.
__global__ __launch_bounds__(32) void gemm_rs_kernel(
    const h16* __restrict__ htatt, const h16* __restrict__ seqT,
    h16* __restrict__ headin, h16* __restrict__ tailin) {
  int nt = blockIdx.x;     // 0..47  (D/16)
  int mt = blockIdx.y;     // 0..7   (P / (16*MT_))
  int b  = blockIdx.z;     // 0..3
  int lane = threadIdx.x;
  const h16* arow[MT_];
#pragma unroll
  for (int q = 0; q < MT_; ++q)
    arow[q] = htatt + (size_t)(b * P_ + mt * (16 * MT_) + q * 16 + (lane & 15)) * C_;
  const h16* bcol = seqT + ((size_t)b * D_ + nt * 16 + (lane & 15)) * C_;
  v8f acc[MT_] = {};
  for (int kb = 0; kb < C_; kb += 32) {
    v16h bf = b_frag(bcol, kb, lane);
#pragma unroll
    for (int q = 0; q < MT_; ++q)
      acc[q] = wmma16(a_frag(arow[q], kb, lane), bf, acc[q]);
  }
  int n = nt * 16 + (lane & 15);
#pragma unroll
  for (int q = 0; q < MT_; ++q) {
    int mbase = mt * (16 * MT_) + q * 16 + ((lane & 16) ? 8 : 0);
#pragma unroll
    for (int r = 0; r < 8; ++r) {
      size_t rg = (size_t)b * P_ + mbase + r;
      h16 v = (h16)acc[q][r];
      headin[rg * KIN_ + (D_ + NT_) + n] = v;
      tailin[rg * KIN_ + (D_ + NT_) + n] = v;
    }
  }
}

// ---------------- WMMA GEMM 2: hs/ts = tanh(X @ W + b) ----------------------
// [2048 x 1792] x [1792 x 768]; blockIdx.z selects head (0) / tail (1).
// MT_ M-tiles per wave -> each weight fragment loaded once per 4 WMMAs.
__global__ __launch_bounds__(32) void gemm_proj_kernel(
    const h16* __restrict__ headin, const h16* __restrict__ tailin,
    const h16* __restrict__ WheadT, const h16* __restrict__ WtailT,
    const float* __restrict__ bhead, const float* __restrict__ btail,
    h16* __restrict__ hs, h16* __restrict__ ts) {
  int nt = blockIdx.x;     // 0..47
  int mt = blockIdx.y;     // 0..31
  int which = blockIdx.z;  // 0 head, 1 tail
  const h16* A = which ? tailin : headin;
  const h16* W = which ? WtailT : WheadT;
  const float* bias = which ? btail : bhead;
  h16* out = which ? ts : hs;
  int lane = threadIdx.x;
  const h16* arow[MT_];
#pragma unroll
  for (int q = 0; q < MT_; ++q)
    arow[q] = A + (size_t)(mt * (16 * MT_) + q * 16 + (lane & 15)) * KIN_;
  const h16* wcol = W + (size_t)(nt * 16 + (lane & 15)) * KIN_;
  v8f acc[MT_] = {};
  for (int kb = 0; kb < KIN_; kb += 32) {
    v16h bf = b_frag(wcol, kb, lane);
#pragma unroll
    for (int q = 0; q < MT_; ++q)
      acc[q] = wmma16(a_frag(arow[q], kb, lane), bf, acc[q]);
  }
  int n = nt * 16 + (lane & 15);
  float bv = bias[n];
#pragma unroll
  for (int q = 0; q < MT_; ++q) {
    int mbase = mt * (16 * MT_) + q * 16 + ((lane & 16) ? 8 : 0);
#pragma unroll
    for (int r = 0; r < 8; ++r)
      out[(size_t)(mbase + r) * EMB_ + n] = (h16)tanhf(acc[q][r] + bv);
  }
}

// ---------------- WMMA GEMM 3: grouped bilinear classifier ------------------
// logits[m,l] = sum_{g,i,j} hs[m,g*64+i]*ts[m,g*64+j]*W[(g*4096+i*64+j), l]
// A-fragments generated in registers (bl never materialized); B from
// pre-transposed f16 W_bil (one 32B contiguous load per WMMA chunk, shared by
// MT_ =4 M-tiles -> 4x less L2 traffic on the 19MB weight stream).
__global__ __launch_bounds__(32) void gemm_bil_kernel(
    const h16* __restrict__ hs, const h16* __restrict__ ts,
    const h16* __restrict__ WbilT, const float* __restrict__ bbil,
    float* __restrict__ logits) {
  int lt = blockIdx.x;     // 0..6 (112/16 label tiles, last partially valid)
  int mt = blockIdx.y;     // 0..31 (2048 / (16*MT_))
  int lane = threadIdx.x;
  const h16* hrow[MT_];
  const h16* trow[MT_];
#pragma unroll
  for (int q = 0; q < MT_; ++q) {
    int row = mt * (16 * MT_) + q * 16 + (lane & 15);
    hrow[q] = hs + (size_t)row * EMB_;
    trow[q] = ts + (size_t)row * EMB_;
  }
  int l = lt * 16 + (lane & 15);
  const h16* wcol = WbilT + (size_t)l * KBIL_;
  v8f acc[MT_] = {};
  for (int g = 0; g < EMB_ / BLK_; ++g) {          // 12 groups
    for (int j0 = 0; j0 < BLK_; j0 += 32) {        // two 32-wide K spans of j
      v16h tseg[MT_];
#pragma unroll
      for (int q = 0; q < MT_; ++q) {
        int toff = g * BLK_ + j0 + ((lane & 16) ? 16 : 0);
        v8h tlo = *(const v8h*)(trow[q] + toff);
        v8h thi = *(const v8h*)(trow[q] + toff + 8);
        tseg[q] = __builtin_shufflevector(tlo, thi, 0, 1, 2, 3, 4, 5, 6, 7,
                                          8, 9, 10, 11, 12, 13, 14, 15);
      }
      int kbgj = g * (BLK_ * BLK_) + j0;
      __builtin_prefetch(wcol + kbgj + 2048, 0, 1);
      for (int ii = 0; ii < BLK_; ii += 8) {
        v8h hv[MT_];
#pragma unroll
        for (int q = 0; q < MT_; ++q)
          hv[q] = *(const v8h*)(hrow[q] + g * BLK_ + ii);
#pragma unroll
        for (int i = 0; i < 8; ++i) {
          int kb = kbgj + (ii + i) * BLK_;
          v16h bf = *(const v16h*)(wcol + kb + ((lane & 16) ? 16 : 0));
#pragma unroll
          for (int q = 0; q < MT_; ++q) {
            h16 hval = hv[q][i];
            v16h a;
#pragma unroll
            for (int m = 0; m < 16; ++m) a[m] = tseg[q][m] * hval;
            acc[q] = wmma16(a, bf, acc[q]);
          }
        }
      }
    }
  }
  if (l < NL_) {                       // guarded stores only; EXEC full at WMMAs
    float bv = bbil[l];
#pragma unroll
    for (int q = 0; q < MT_; ++q) {
      int mbase = mt * (16 * MT_) + q * 16 + ((lane & 16) ? 8 : 0);
#pragma unroll
      for (int r = 0; r < 8; ++r)
        logits[(size_t)(mbase + r) * NL_ + l] = acc[q][r] + bv;
    }
  }
}

// ---------------- host launcher ---------------------------------------------

extern "C" void kernel_launch(void* const* d_in, const int* in_sizes, int n_in,
                              void* d_out, int out_size, void* d_ws, size_t ws_size,
                              hipStream_t stream) {
  const float* seq        = (const float*)d_in[0];
  const float* attention  = (const float*)d_in[1];
  const int*   entity_pos = (const int*)d_in[2];
  const int*   sent_pos   = (const int*)d_in[3];
  const int*   virtual_pos= (const int*)d_in[4];
  const int*   hts        = (const int*)d_in[5];
  const float* node_type  = (const float*)d_in[6];
  const float* W_head     = (const float*)d_in[7];
  const float* b_head     = (const float*)d_in[8];
  const float* W_tail     = (const float*)d_in[9];
  const float* b_tail     = (const float*)d_in[10];
  const float* W_bil      = (const float*)d_in[11];
  const float* b_bil      = (const float*)d_in[12];

  // output layout: logits | mention | sent | virtual (f32, flat, return order)
  float* out_logits  = (float*)d_out;
  float* out_mention = out_logits + (size_t)B_ * P_ * NL_;                  // 198656
  float* out_sent    = out_mention + (size_t)B_ * E_ * K_ * D_;             // +196608
  float* out_virt    = out_sent + (size_t)B_ * S_ * D_;                     // +73728

  // workspace carve-up (256B aligned regions)
  char* ws = (char*)d_ws;
  size_t off = 0;
  auto carve = [&](size_t bytes) {
    char* p = ws + off;
    off += (bytes + 255) & ~(size_t)255;
    return p;
  };
  float* e_att    = (float*)carve((size_t)B_ * E_ * H_ * C_ * sizeof(float));
  h16*   ent_hid  = (h16*)carve((size_t)B_ * E_ * (D_ + NT_) * sizeof(h16));
  h16*   htatt    = (h16*)carve((size_t)B_ * P_ * C_ * sizeof(h16));
  h16*   seqT     = (h16*)carve((size_t)B_ * D_ * C_ * sizeof(h16));
  h16*   headin   = (h16*)carve((size_t)B_ * P_ * KIN_ * sizeof(h16));
  h16*   tailin   = (h16*)carve((size_t)B_ * P_ * KIN_ * sizeof(h16));
  h16*   WheadT   = (h16*)carve((size_t)EMB_ * KIN_ * sizeof(h16));
  h16*   WtailT   = (h16*)carve((size_t)EMB_ * KIN_ * sizeof(h16));
  h16*   WbilT    = (h16*)carve((size_t)NLPAD_ * KBIL_ * sizeof(h16));
  h16*   hs_h     = (h16*)carve((size_t)B_ * P_ * EMB_ * sizeof(h16));
  h16*   ts_h     = (h16*)carve((size_t)B_ * P_ * EMB_ * sizeof(h16));
  (void)ws_size; (void)in_sizes; (void)n_in; (void)out_size;

  // 1) weight f16 transposes
  transpose_w_kernel<<<2048, 256, 0, stream>>>(W_head, WheadT, KIN_, EMB_, EMB_);
  transpose_w_kernel<<<2048, 256, 0, stream>>>(W_tail, WtailT, KIN_, EMB_, EMB_);
  transpose_w_kernel<<<4096, 256, 0, stream>>>(W_bil, WbilT, KBIL_, NL_, NLPAD_);

  // 2) seq^T f16
  seq_transpose_kernel<<<4096, 256, 0, stream>>>(seq, seqT);

  // 3) row gathers (mention/sent/virtual) directly into d_out
  gather_kernel<<<B_ * E_ * K_ + B_ * S_ + B_ * V_, 256, 0, stream>>>(
      seq, entity_pos, sent_pos, virtual_pos, out_mention, out_sent, out_virt);

  // 4) entity logsumexp + node-type concat (reads mention rows from d_out)
  entity_kernel<<<B_ * E_, 256, 0, stream>>>(out_mention, node_type, ent_hid);

  // 5) mention-averaged attention rows
  eatt_kernel<<<B_ * E_ * H_, 256, 0, stream>>>(attention, entity_pos, e_att);

  // 6) pair attention product -> normalized ht_att (f16)
  htatt_kernel<<<B_ * P_, 256, 0, stream>>>(e_att, hts, htatt);

  // 7) pair entity gathers into concat buffers [0:1024)
  pairgather_kernel<<<B_ * P_, 256, 0, stream>>>(ent_hid, hts, headin, tailin);

  // 8) WMMA GEMM: rs = ht_att @ seq  -> concat buffers [1024:1792)
  gemm_rs_kernel<<<dim3(D_ / 16, P_ / (16 * MT_), B_), 32, 0, stream>>>(
      htatt, seqT, headin, tailin);

  // 9) WMMA GEMM: hs/ts = tanh(concat @ W + b)
  gemm_proj_kernel<<<dim3(EMB_ / 16, (B_ * P_) / (16 * MT_), 2), 32, 0, stream>>>(
      headin, tailin, WheadT, WtailT, b_head, b_tail, hs_h, ts_h);

  // 10) WMMA GEMM: grouped bilinear -> logits
  gemm_bil_kernel<<<dim3(NLPAD_ / 16, (B_ * P_) / (16 * MT_)), 32, 0, stream>>>(
      hs_h, ts_h, WbilT, b_bil, out_logits);
}
